// MLHGT_79044578115935
// MI455X (gfx1250) — matrix operand order repaired and checked by reading, not compiled, not run on hardware
//
#include <hip/hip_runtime.h>
#include <hip/hip_bf16.h>
#include <math.h>

// ---------------------------------------------------------------------------
// Types for CDNA5 WMMA (wave32, 16x16x32 bf16 -> f32)
// ---------------------------------------------------------------------------
typedef __bf16 bf16;
typedef __attribute__((ext_vector_type(16))) __bf16 v16bf;
typedef __attribute__((ext_vector_type(8)))  __bf16 v8bf;
typedef __attribute__((ext_vector_type(8)))  float  v8f;

#define HID 256
#define NHEAD 4
#define DHEAD 64
#define NLAYERS 3

// --------------------------- device helpers --------------------------------
__device__ __forceinline__ bf16 to_bf16(float f) {
  unsigned u = __float_as_uint(f);
  unsigned r = u + 0x7FFFu + ((u >> 16) & 1u);   // round to nearest even
  union { unsigned short s; bf16 b; } cv;
  cv.s = (unsigned short)(r >> 16);
  return cv.b;
}
__device__ __forceinline__ float from_bf16(bf16 b) {
  union { unsigned short s; bf16 b; } cv;
  cv.b = b;
  return __uint_as_float(((unsigned)cv.s) << 16);
}
// order-preserving float->uint key for atomicMax-based segment max
__device__ __forceinline__ unsigned fkey(float x) {
  unsigned u = __float_as_uint(x);
  return (u & 0x80000000u) ? ~u : (u | 0x80000000u);
}
__device__ __forceinline__ float fdecode(unsigned k) {
  unsigned u = (k & 0x80000000u) ? (k & 0x7fffffffu) : ~k;
  return __uint_as_float(u);
}
__device__ __forceinline__ v16bf make_frag(v8bf lo, v8bf hi) {
  v16bf f;
#pragma unroll
  for (int i = 0; i < 8; ++i) { f[i] = lo[i]; f[i + 8] = hi[i]; }
  return f;
}

// ---------------------------------------------------------------------------
// bf16 WMMA GEMM:  Out[n, o] = sum_k A[n,k] * W[o,k] + bias[o]
//   MROWS = 16-row A tiles per wave, TPW = 16-col tiles per wave.
//   KC    = compile-time K (fully unrolled, double-buffered fragment loads).
//   MODE  = 0: store bf16 output; 2: H = sigmoid(skip)*(acc+b)+(1-s)*H
//           with bf16 mirror refresh.
//   gridDim.y = batch (per-head relation transforms).
// Block = 128 threads = 4 waves; block covers 16*MROWS rows x 64*TPW cols.
// ---------------------------------------------------------------------------
template <int MROWS, int TPW, int MODE, int KC>
__global__ __launch_bounds__(128)
void MLHGT_wmma_gemm(const bf16* __restrict__ A, long long aBS, int lda,
                     const bf16* __restrict__ W, long long wBS, int ldw,
                     const float* __restrict__ bias,
                     int Nrows,
                     bf16* __restrict__ Obf, long long oBS, int ldo,
                     const float* __restrict__ skipParam,
                     float* __restrict__ Hf32, bf16* __restrict__ Hbf) {
  A += (long long)blockIdx.y * aBS;
  W += (long long)blockIdx.y * wBS;
  const long long obase = (long long)blockIdx.y * oBS;

  const int lane = threadIdx.x & 31;
  const int wv   = threadIdx.x >> 5;     // wave id 0..3
  const int g    = lane >> 4;            // lane group 0/1
  const int ln16 = lane & 15;

  const int row0 = blockIdx.x * (16 * MROWS);
  const bf16* Arow[MROWS];
#pragma unroll
  for (int m = 0; m < MROWS; ++m) {
    int r = row0 + m * 16 + ln16;
    if (r >= Nrows) r = Nrows - 1;       // clamp (stores are guarded)
    Arow[m] = A + (long long)r * lda;
  }
  const int col0 = wv * (TPW * 16);
  const bf16* Wr[TPW];
#pragma unroll
  for (int t = 0; t < TPW; ++t)
    Wr[t] = W + (long long)(col0 + t * 16 + ln16) * ldw + 16 * g;

  v8f acc[MROWS][TPW];
#pragma unroll
  for (int m = 0; m < MROWS; ++m)
#pragma unroll
    for (int t = 0; t < TPW; ++t) acc[m][t] = v8f{};

  constexpr int NK = KC / 32;
  v16bf af[2][MROWS];
  v16bf bfr[2][TPW];

  // fragment loads for K-step i into buffer slot s
  auto loadStep = [&](int i, int s) {
    const int k0 = i * 32;
#pragma unroll
    for (int m = 0; m < MROWS; ++m) {
      v8bf alo = *(const v8bf*)(Arow[m] + k0 + 16 * g);
      v8bf ahi = *(const v8bf*)(Arow[m] + k0 + 16 * g + 8);
      af[s][m] = make_frag(alo, ahi);
    }
#pragma unroll
    for (int t = 0; t < TPW; ++t) {
      v8bf blo = *(const v8bf*)(Wr[t] + k0);
      v8bf bhi = *(const v8bf*)(Wr[t] + k0 + 8);
      bfr[s][t] = make_frag(blo, bhi);
    }
  };

  loadStep(0, 0);
#pragma unroll
  for (int i = 0; i < NK; ++i) {
    const int cur = i & 1, nxt = cur ^ 1;
    if (i + 2 < NK) __builtin_prefetch(Arow[0] + (i + 2) * 32, 0, 1);
    if (i + 1 < NK) loadStep(i + 1, nxt);  // overlap loads with WMMA issue
#pragma unroll
    for (int m = 0; m < MROWS; ++m)
#pragma unroll
      for (int t = 0; t < TPW; ++t)
        acc[m][t] = __builtin_amdgcn_wmma_f32_16x16x32_bf16(
            false, af[cur][m], false, bfr[cur][t], (short)0, acc[m][t], false, false);
  }

  float sa = 0.f, sb = 0.f;
  if constexpr (MODE == 2) {
    sa = 1.f / (1.f + __expf(-skipParam[0]));
    sb = 1.f - sa;
  }
#pragma unroll
  for (int m = 0; m < MROWS; ++m) {
#pragma unroll
    for (int t = 0; t < TPW; ++t) {
      const int col = col0 + t * 16 + ln16;
      const float bv = bias ? bias[col] : 0.f;
#pragma unroll
      for (int r = 0; r < 8; ++r) {
        const int row = row0 + m * 16 + r + 8 * g;  // D layout: M = r + 8*(lane/16)
        if (row < Nrows) {
          const long long idx = obase + (long long)row * ldo + col;
          float v = acc[m][t][r] + bv;
          if constexpr (MODE == 0) {
            Obf[idx] = to_bf16(v);
          } else {
            float hn = sa * v + sb * Hf32[idx];
            Hf32[idx] = hn;
            Hbf[idx]  = to_bf16(hn);
          }
        }
      }
    }
  }
}

// --------------------------- elementwise kernels ---------------------------
__global__ void MLHGT_fill_u32(unsigned* __restrict__ p, unsigned v, long long n) {
  long long i = (long long)blockIdx.x * blockDim.x + threadIdx.x;
  if (i < n) p[i] = v;
}
__global__ void MLHGT_f32_to_bf16(const float* __restrict__ s, bf16* __restrict__ d,
                                  long long n) {
  long long i = (long long)blockIdx.x * blockDim.x + threadIdx.x;
  if (i < n) d[i] = to_bf16(s[i]);
}
__global__ void MLHGT_gelu_to_bf16(const float* __restrict__ s, bf16* __restrict__ d,
                                   long long n) {
  long long i = (long long)blockIdx.x * blockDim.x + threadIdx.x;
  if (i < n) {
    float x = s[i];
    d[i] = to_bf16(0.5f * x * (1.f + erff(x * 0.70710678118f)));
  }
}
// a[h][d][e] (f32) -> at[h][e][d] (bf16), so B-fragments read contiguous K
__global__ void MLHGT_transpose_rel(const float* __restrict__ a, bf16* __restrict__ at) {
  int i = blockIdx.x * blockDim.x + threadIdx.x;
  if (i >= NHEAD * DHEAD * DHEAD) return;
  int h = i >> 12, r = i & 4095, e = r >> 6, dd = r & 63;
  at[(h << 12) + (e << 6) + dd] = to_bf16(a[(h << 12) + (dd << 6) + e]);
}

// --------------------------- node embedding --------------------------------
__global__ void MLHGT_embed_lig(const float* __restrict__ x, const float* __restrict__ emb,
                                const float* __restrict__ posw,
                                float* __restrict__ h, bf16* __restrict__ hb, int NL) {
  long long i = (long long)blockIdx.x * blockDim.x + threadIdx.x;
  if (i >= (long long)NL * 128) return;
  int n = (int)(i >> 7), j = (int)(i & 127);
  const float* xr = x + (long long)n * 4;
  int t = (int)xr[0];
  float lo = emb[t * 128 + j];
  float hi = posw[j * 3 + 0] * xr[1] + posw[j * 3 + 1] * xr[2] + posw[j * 3 + 2] * xr[3];
  long long o = (long long)n * HID;
  h[o + j] = lo;          h[o + 128 + j] = hi;
  hb[o + j] = to_bf16(lo); hb[o + 128 + j] = to_bf16(hi);
}
__global__ void MLHGT_embed_met(const float* __restrict__ x, const float* __restrict__ emb,
                                const float* __restrict__ metw, const float* __restrict__ posw,
                                float* __restrict__ h, bf16* __restrict__ hb, int NM) {
  long long i = (long long)blockIdx.x * blockDim.x + threadIdx.x;
  if (i >= (long long)NM * 128) return;
  int n = (int)(i >> 7), j = (int)(i & 127);
  const float* xr = x + (long long)n * 21;
  int t = (int)xr[17];
  float lo = emb[t * 128 + j];
#pragma unroll
  for (int c = 0; c < 17; ++c) lo += metw[j * 17 + c] * xr[c];
  float hi = posw[j * 3 + 0] * xr[18] + posw[j * 3 + 1] * xr[19] + posw[j * 3 + 2] * xr[20];
  long long o = (long long)n * HID;
  h[o + j] = lo;          h[o + 128 + j] = hi;
  hb[o + j] = to_bf16(lo); hb[o + 128 + j] = to_bf16(hi);
}

// --------------------------- edge attention --------------------------------
__global__ void MLHGT_attn_logits(const int* __restrict__ src, const int* __restrict__ dst,
                                  const bf16* __restrict__ q, const bf16* __restrict__ kk,
                                  const float* __restrict__ prel,
                                  float* __restrict__ alpha, unsigned* __restrict__ mkey,
                                  int E) {
  long long i = (long long)blockIdx.x * blockDim.x + threadIdx.x;
  if (i >= (long long)E * NHEAD) return;
  int e = (int)(i >> 2), h = (int)(i & 3);
  int s = src[e], d = dst[e];
  const bf16* qp = q  + (long long)d * HID + h * DHEAD;
  const bf16* kp = kk + (long long)s * HID + h * DHEAD;
  float acc = 0.f;
#pragma unroll 8
  for (int j = 0; j < DHEAD; ++j) acc += from_bf16(qp[j]) * from_bf16(kp[j]);
  float lg = acc * prel[h] * 0.125f;    // 1/sqrt(64)
  alpha[i] = lg;
  atomicMax(&mkey[d * NHEAD + h], fkey(lg));
}
__global__ void MLHGT_attn_expsum(const int* __restrict__ dst, float* __restrict__ alpha,
                                  const unsigned* __restrict__ mkey,
                                  float* __restrict__ ssum, int E) {
  long long i = (long long)blockIdx.x * blockDim.x + threadIdx.x;
  if (i >= (long long)E * NHEAD) return;
  int e = (int)(i >> 2), h = (int)(i & 3);
  int d = dst[e];
  float m = fdecode(mkey[d * NHEAD + h]);
  float ex = __expf(alpha[i] - m);
  alpha[i] = ex;
  atomicAdd(&ssum[d * NHEAD + h], ex);
}
__global__ void MLHGT_attn_aggregate(const int* __restrict__ src, const int* __restrict__ dst,
                                     const bf16* __restrict__ vv,
                                     const float* __restrict__ alpha,
                                     const float* __restrict__ ssum,
                                     float* __restrict__ out, int E) {
  long long i = (long long)blockIdx.x * blockDim.x + threadIdx.x;
  if (i >= (long long)E * NHEAD * 4) return;
  int e  = (int)(i >> 4);
  int h  = (int)((i >> 2) & 3);
  int qd = (int)(i & 3);
  int s = src[e], d = dst[e];
  float w = alpha[((long long)e << 2) + h] / (ssum[d * NHEAD + h] + 1e-16f);
  const bf16* vp = vv  + (long long)s * HID + h * DHEAD + qd * 16;
  float*      op = out + (long long)d * HID + h * DHEAD + qd * 16;
#pragma unroll
  for (int j = 0; j < 16; ++j) atomicAdd(&op[j], from_bf16(vp[j]) * w);
}

// --------------------------- pooling + head --------------------------------
__global__ void MLHGT_pool_add(const float* __restrict__ h, const int* __restrict__ batch,
                               float* __restrict__ pool, float* __restrict__ cnt,
                               int N, int colOff) {
  long long i = (long long)blockIdx.x * blockDim.x + threadIdx.x;
  if (i >= (long long)N * HID) return;
  int n = (int)(i >> 8), j = (int)(i & 255);
  int b = batch[n];
  atomicAdd(&pool[(long long)b * 512 + colOff + j], h[(long long)n * HID + j]);
  if (j == 0) atomicAdd(&cnt[b], 1.f);
}
__global__ __launch_bounds__(256)
void MLHGT_head(const float* __restrict__ pool, const float* __restrict__ cntL,
                const float* __restrict__ cntM,
                const float* __restrict__ w1, const float* __restrict__ b1,
                const float* __restrict__ w2, const float* __restrict__ b2,
                float* __restrict__ out, int B) {
  int gIdx = blockIdx.x;
  int o = threadIdx.x;
  __shared__ float z[512];
  __shared__ float red[256];
  float cl = fmaxf(cntL[gIdx], 1.f), cm = fmaxf(cntM[gIdx], 1.f);
  for (int j = o; j < 512; j += 256)
    z[j] = pool[(long long)gIdx * 512 + j] / (j < 256 ? cl : cm);
  __syncthreads();
  float acc = b1[o];
  for (int j = 0; j < 512; ++j) acc += z[j] * w1[o * 512 + j];
  float hp = (acc > 20.f) ? acc : log1pf(__expf(acc));   // softplus
  red[o] = hp * w2[o];
  __syncthreads();
  for (int sft = 128; sft > 0; sft >>= 1) {
    if (o < sft) red[o] += red[o + sft];
    __syncthreads();
  }
  if (o == 0) out[gIdx] = red[0] + b2[0];
}

// ---------------------------------------------------------------------------
// Host-side orchestration. Params follow JAX pytree (sorted-dict-key) order.
// ---------------------------------------------------------------------------
enum {
  IN_XLIG = 0, IN_XMET, IN_EBOND, IN_EDONOR, IN_EBACK, IN_BATL, IN_BATM, IN_NG,
  P_EMB, P_HB1, P_HB2, P_HW1, P_HW2,
  P_LAYER0                                   // = 13
};
enum {
  L_BACK_A = 0, L_BACK_M, L_BACK_P,
  L_BOND_A, L_BOND_M, L_BOND_P,
  L_DONOR_A, L_DONOR_M, L_DONOR_P,
  L_LIG_AB, L_LIG_AW, L_LIG_KB, L_LIG_KW, L_LIG_QB, L_LIG_QW, L_LIG_SKIP, L_LIG_VB, L_LIG_VW,
  L_MET_AB, L_MET_AW, L_MET_KB, L_MET_KW, L_MET_QB, L_MET_QW, L_MET_SKIP, L_MET_VB, L_MET_VW,
  L_COUNT                                    // = 27
};
#define P_METW (P_LAYER0 + NLAYERS * L_COUNT)  // 94
#define P_POSW (P_METW + 1)                    // 95

static inline unsigned nblk(long long n, int t) { return (unsigned)((n + t - 1) / t); }

extern "C" void kernel_launch(void* const* d_in, const int* in_sizes, int n_in,
                              void* d_out, int out_size, void* d_ws, size_t ws_size,
                              hipStream_t stream) {
  (void)n_in; (void)ws_size;
  const float* x_lig = (const float*)d_in[IN_XLIG];
  const float* x_met = (const float*)d_in[IN_XMET];
  const int* e_bond  = (const int*)d_in[IN_EBOND];
  const int* e_donor = (const int*)d_in[IN_EDONOR];
  const int* e_back  = (const int*)d_in[IN_EBACK];
  const int* bat_l   = (const int*)d_in[IN_BATL];
  const int* bat_m   = (const int*)d_in[IN_BATM];

  const int NL = in_sizes[IN_XLIG] / 4;
  const int NM = in_sizes[IN_XMET] / 21;
  const int EB = in_sizes[IN_EBOND] / 2;
  const int ED = in_sizes[IN_EDONOR] / 2;
  const int B  = out_size;

  // ---- carve workspace ----
  char* base = (char*)d_ws;
  size_t off = 0;
  auto carve = [&](size_t bytes) -> char* {
    char* p = base + off;
    off += (bytes + 255) & ~(size_t)255;
    return p;
  };
  const size_t NLH = (size_t)NL * HID, NMH = (size_t)NM * HID;
  float* h_lig  = (float*)carve(NLH * 4);
  float* h_met  = (float*)carve(NMH * 4);
  bf16*  hb_lig = (bf16*)carve(NLH * 2);
  bf16*  hb_met = (bf16*)carve(NMH * 2);
  bf16*  k_l = (bf16*)carve(NLH * 2);
  bf16*  q_l = (bf16*)carve(NLH * 2);
  bf16*  v_l = (bf16*)carve(NLH * 2);
  bf16*  k_m = (bf16*)carve(NMH * 2);
  bf16*  q_m = (bf16*)carve(NMH * 2);
  bf16*  v_m = (bf16*)carve(NMH * 2);
  bf16*  kk  = (bf16*)carve(NLH * 2);     // sized for max source set
  bf16*  vv  = (bf16*)carve(NLH * 2);
  float* out_l = (float*)carve(NLH * 4);
  float* out_m = (float*)carve(NMH * 4);
  float* alpha = (float*)carve((size_t)EB * NHEAD * 4);
  unsigned* mkey = (unsigned*)carve((size_t)NL * NHEAD * 4);
  float* ssum = (float*)carve((size_t)NL * NHEAD * 4);
  bf16*  gbuf = (bf16*)carve(NLH * 2);
  bf16*  wslot[8];
  for (int i = 0; i < 8; ++i) wslot[i] = (bf16*)carve((size_t)HID * HID * 2);
  bf16*  aT = (bf16*)carve((size_t)NHEAD * DHEAD * DHEAD * 2);
  bf16*  mT = (bf16*)carve((size_t)NHEAD * DHEAD * DHEAD * 2);
  float* pool = (float*)carve((size_t)B * 512 * 4);
  float* cntL = (float*)carve((size_t)B * 4);
  float* cntM = (float*)carve((size_t)B * 4);

  auto zero = [&](void* p, long long words) {
    MLHGT_fill_u32<<<nblk(words, 256), 256, 0, stream>>>((unsigned*)p, 0u, words);
  };
  // Ncols=256 projection GEMM (32 rows/block), bf16 output
  auto gemm256 = [&](const bf16* A, const bf16* W, const float* bias, int Nr, bf16* Ob) {
    dim3 grid((Nr + 31) / 32, 1);
    MLHGT_wmma_gemm<2, 4, 0, 256><<<grid, 128, 0, stream>>>(
        A, 0, HID, W, 0, HID, bias, Nr, Ob, 0, HID, nullptr, nullptr, nullptr);
  };
  // Ncols=256 output GEMM with fused sigmoid-skip combine
  auto gemmSkip = [&](const bf16* A, const bf16* W, const float* bias, int Nr,
                      const float* skip, float* Hf, bf16* Hb) {
    dim3 grid((Nr + 31) / 32, 1);
    MLHGT_wmma_gemm<2, 4, 2, 256><<<grid, 128, 0, stream>>>(
        A, 0, HID, W, 0, HID, bias, Nr, nullptr, 0, HID, skip, Hf, Hb);
  };
  // per-head 64x64 transform (batched over heads in gridDim.y)
  auto gemmHead = [&](const bf16* A, const bf16* W, int Nr, bf16* Ob) {
    dim3 grid((Nr + 15) / 16, NHEAD);
    MLHGT_wmma_gemm<1, 1, 0, 64><<<grid, 128, 0, stream>>>(
        A, DHEAD, HID, W, DHEAD * DHEAD, DHEAD, nullptr, Nr,
        Ob, DHEAD, HID, nullptr, nullptr, nullptr);
  };
  // one full relation: per-head K/V transform (WMMA), softmax, scatter-aggregate
  auto relation = [&](const bf16* ksrc, const bf16* vsrc, const bf16* qdst,
                      const float* a_rel, const float* m_rel, const float* p_rel,
                      const int* src, const int* dst, int E, int Nsrc, int Ndst,
                      float* outDst) {
    MLHGT_transpose_rel<<<nblk(NHEAD * DHEAD * DHEAD, 256), 256, 0, stream>>>(a_rel, aT);
    MLHGT_transpose_rel<<<nblk(NHEAD * DHEAD * DHEAD, 256), 256, 0, stream>>>(m_rel, mT);
    gemmHead(ksrc, aT, Nsrc, kk);
    gemmHead(vsrc, mT, Nsrc, vv);
    zero(mkey, (long long)Ndst * NHEAD);
    zero(ssum, (long long)Ndst * NHEAD);
    MLHGT_attn_logits<<<nblk((long long)E * NHEAD, 256), 256, 0, stream>>>(
        src, dst, qdst, kk, p_rel, alpha, mkey, E);
    MLHGT_attn_expsum<<<nblk((long long)E * NHEAD, 256), 256, 0, stream>>>(
        dst, alpha, mkey, ssum, E);
    MLHGT_attn_aggregate<<<nblk((long long)E * NHEAD * 4, 256), 256, 0, stream>>>(
        src, dst, vv, alpha, ssum, outDst, E);
  };

  // ---- initial embeddings ----
  MLHGT_embed_lig<<<nblk((long long)NL * 128, 256), 256, 0, stream>>>(
      x_lig, (const float*)d_in[P_EMB], (const float*)d_in[P_POSW], h_lig, hb_lig, NL);
  MLHGT_embed_met<<<nblk((long long)NM * 128, 256), 256, 0, stream>>>(
      x_met, (const float*)d_in[P_EMB], (const float*)d_in[P_METW],
      (const float*)d_in[P_POSW], h_met, hb_met, NM);

  // ---- HGT layers ----
  for (int l = 0; l < NLAYERS; ++l) {
    const int LB = P_LAYER0 + l * L_COUNT;
    auto P = [&](int o) -> const float* { return (const float*)d_in[LB + o]; };

    const int widx[8] = {L_LIG_KW, L_LIG_QW, L_LIG_VW, L_LIG_AW,
                         L_MET_KW, L_MET_QW, L_MET_VW, L_MET_AW};
    for (int i = 0; i < 8; ++i)
      MLHGT_f32_to_bf16<<<nblk(HID * HID, 256), 256, 0, stream>>>(
          P(widx[i]), wslot[i], (long long)HID * HID);

    // K/Q/V projections (WMMA, bf16 in/out, f32 accumulate)
    gemm256(hb_lig, wslot[0], P(L_LIG_KB), NL, k_l);
    gemm256(hb_lig, wslot[1], P(L_LIG_QB), NL, q_l);
    gemm256(hb_lig, wslot[2], P(L_LIG_VB), NL, v_l);
    gemm256(hb_met, wslot[4], P(L_MET_KB), NM, k_m);
    gemm256(hb_met, wslot[5], P(L_MET_QB), NM, q_m);
    gemm256(hb_met, wslot[6], P(L_MET_VB), NM, v_m);

    zero(out_l, (long long)NLH);
    zero(out_m, (long long)NMH);

    // (ligand, bond, ligand) + (metal, backbonding, ligand) summed into out_l
    relation(k_l, v_l, q_l, P(L_BOND_A), P(L_BOND_M), P(L_BOND_P),
             e_bond, e_bond + EB, EB, NL, NL, out_l);
    relation(k_m, v_m, q_l, P(L_BACK_A), P(L_BACK_M), P(L_BACK_P),
             e_back, e_back + ED, ED, NM, NL, out_l);
    // (ligand, donor, metal) into out_m
    relation(k_l, v_l, q_m, P(L_DONOR_A), P(L_DONOR_M), P(L_DONOR_P),
             e_donor, e_donor + ED, ED, NL, NM, out_m);

    // update: gelu -> linear (WMMA) -> sigmoid-skip combine, fused in epilogue
    MLHGT_gelu_to_bf16<<<nblk((long long)NLH, 256), 256, 0, stream>>>(out_l, gbuf, NLH);
    gemmSkip(gbuf, wslot[3], P(L_LIG_AB), NL, P(L_LIG_SKIP), h_lig, hb_lig);
    MLHGT_gelu_to_bf16<<<nblk((long long)NMH, 256), 256, 0, stream>>>(out_m, gbuf, NMH);
    gemmSkip(gbuf, wslot[7], P(L_MET_AB), NM, P(L_MET_SKIP), h_met, hb_met);
  }

  // ---- mean pool + MLP head ----
  zero(pool, (long long)B * 512);
  zero(cntL, B);
  zero(cntM, B);
  MLHGT_pool_add<<<nblk((long long)NLH, 256), 256, 0, stream>>>(h_lig, bat_l, pool, cntL, NL, 0);
  MLHGT_pool_add<<<nblk((long long)NMH, 256), 256, 0, stream>>>(h_met, bat_m, pool, cntM, NM, 256);
  MLHGT_head<<<B, 256, 0, stream>>>(pool, cntL, cntM,
                                    (const float*)d_in[P_HW1], (const float*)d_in[P_HB1],
                                    (const float*)d_in[P_HW2], (const float*)d_in[P_HB2],
                                    (float*)d_out, B);
}
